// HNNF_76802605187179
// MI455X (gfx1250) — matrix operand order repaired and checked by reading, not compiled, not run on hardware
//
#include <hip/hip_runtime.h>

// ---------------------------------------------------------------------------
// MI455X / gfx1250, wave32. Dominant cost: [4096x8192] @ [8192x8192]^T GEMM
// (550 GFLOP). fp32 emulated via fp16 hi/lo split (3x v_wmma_f32_16x16x32_f16
// per K=32 slab, f32 accumulate). Double-buffered LDS pipeline; A-operand
// staged with CDNA5 async-to-LDS copies (ASYNCcnt-tracked, bypasses VGPRs).
// ---------------------------------------------------------------------------

typedef __attribute__((ext_vector_type(16))) _Float16     v16h;
typedef __attribute__((ext_vector_type(8)))  float        v8f;
typedef __attribute__((ext_vector_type(4)))  float        f32x4;
typedef __attribute__((ext_vector_type(4)))  unsigned int u32x4;
typedef __attribute__((ext_vector_type(2)))  unsigned int u32x2;

union Frag  { v16h v; u32x4 q[2]; };
union Half4 { _Float16 h[4]; u32x2 u; };

#define B_ROWS   4096
#define KDIM     8192
#define NDIM     8192

#if __has_builtin(__builtin_amdgcn_global_load_async_to_lds_b128)
#define HAVE_ASYNC_LDS 1
#else
#define HAVE_ASYNC_LDS 0
#endif
#if __has_builtin(__builtin_amdgcn_s_wait_asynccnt)
#define HAVE_WAIT_ASYNC 1
#else
#define HAVE_WAIT_ASYNC 0
#endif

// Builtin signature (from round-2 diagnostic): param0 is gcc-vector
// 'int __vector(4)' in the global address space; param1 is the LDS dest.
typedef int v4i_gcc __attribute__((vector_size(16)));
typedef __attribute__((address_space(1))) v4i_gcc* gptr128_t;
typedef __attribute__((address_space(3))) v4i_gcc* lptr128_t;

__device__ __forceinline__ void async_copy16(_Float16* lds, const _Float16* g) {
#if HAVE_ASYNC_LDS
    __builtin_amdgcn_global_load_async_to_lds_b128(
        (gptr128_t)g, (lptr128_t)lds, 0, 0);
#else
    *(u32x4*)lds = *(const u32x4*)g;
#endif
}

__device__ __forceinline__ void wait_async0() {
#if HAVE_ASYNC_LDS
#if HAVE_WAIT_ASYNC
    __builtin_amdgcn_s_wait_asynccnt(0);
#else
    asm volatile("s_wait_asynccnt 0" ::: "memory");
#endif
#endif
}

// ============================================================================
// Kernel 1: per-row feature tree + h1 = feats @ Wf1^T + bf1, written pre-split
// as f16 hi/lo (the "A" operand of the big GEMM). One block per batch row.
// ============================================================================
__global__ __launch_bounds__(256) void hnnf_feats_h1(
    const float* __restrict__ x0, const float* __restrict__ x1,
    const float* __restrict__ x2, const float* __restrict__ x3,
    const float* __restrict__ W0, const float* __restrict__ b0,
    const float* __restrict__ W1, const float* __restrict__ b1,
    const float* __restrict__ W2, const float* __restrict__ b2,
    const float* __restrict__ Wf1, const float* __restrict__ bf1,
    _Float16* __restrict__ Ahi, _Float16* __restrict__ Alo)
{
    __shared__ float xs[8192];   // x0 row, staged coalesced (32 KB)
    __shared__ float t0[256];    // level-0 node outputs
    __shared__ float u[32];      // u[0:16]=level1(x0 path), u[16:32]=level1(x1)
    __shared__ float fs[4];      // feats

    const int b = blockIdx.x;
    const int t = threadIdx.x;

    {
        const f32x4* g   = (const f32x4*)(x0 + (size_t)b * 8192);
        f32x4*       dst = (f32x4*)xs;
        #pragma unroll
        for (int i = 0; i < 8; ++i) dst[t + i * 256] = g[t + i * 256];
    }
    __syncthreads();

    {   // level 0: 256 nodes, dot-32 + bias + relu
        float s = 0.f;
        const float* xr = &xs[t * 32];
        const float* wr = &W0[t * 32];
        #pragma unroll
        for (int k = 0; k < 32; ++k) s += xr[k] * wr[k];
        t0[t] = fmaxf(s + b0[t], 0.f);
    }
    __syncthreads();

    if (t < 32) {   // level 1 over t0 (x0 path) and over x1 row
        const int i = t & 15;
        const float* src = (t < 16) ? &t0[i * 16] : (x1 + (size_t)b * 256 + i * 16);
        float s = 0.f;
        #pragma unroll
        for (int k = 0; k < 16; ++k) s += src[k] * W1[i * 16 + k];
        u[t] = fmaxf(s + b1[i], 0.f);
    }
    __syncthreads();

    if (t < 4) {    // level 2 (single node) -> feats[4]
        float v;
        if (t == 3) {
            v = x3[b];
        } else {
            float s = 0.f;
            if (t == 0) {
                #pragma unroll
                for (int k = 0; k < 16; ++k) s += u[k] * W2[k];
            } else if (t == 1) {
                #pragma unroll
                for (int k = 0; k < 16; ++k) s += u[16 + k] * W2[k];
            } else {
                const float* xr = x2 + (size_t)b * 16;
                #pragma unroll
                for (int k = 0; k < 16; ++k) s += xr[k] * W2[k];
            }
            v = fmaxf(s + b2[0], 0.f);
        }
        fs[t] = v;
    }
    __syncthreads();

    const float f0 = fs[0], f1 = fs[1], f2 = fs[2], f3 = fs[3];
    const size_t rowbase = (size_t)b * KDIM;
    for (int c = 0; c < 32; ++c) {
        const int o = c * 256 + t;
        const f32x4 w = ((const f32x4*)Wf1)[o];
        const float h = f0 * w.x + f1 * w.y + f2 * w.z + f3 * w.w + bf1[o];
        const _Float16 hi = (_Float16)h;
        const float    lo = h - (float)hi;
        Ahi[rowbase + o] = hi;
        Alo[rowbase + o] = (_Float16)lo;
    }
}

// ============================================================================
// Kernel 2: out = | Ahi/lo-GEMM-Wf2^T + bf2 |   (M=4096, N=8192, K=8192)
//   BM=BN=128, BK=32, 8 waves x (64x32). fp16x3: Alo*Bhi + Ahi*Blo + Ahi*Bhi.
//   Double-buffered LDS; A via async-to-LDS, B fp32 loads early / split late.
// ============================================================================
#define LDT 40            // LDS tile row stride in halves (80 B: 64 data + pad)
#define KSTEPS (KDIM / 32)

__global__ __launch_bounds__(256) void hnnf_gemm(
    const _Float16* __restrict__ Ahi, const _Float16* __restrict__ Alo,
    const float* __restrict__ Wf2, const float* __restrict__ bf2,
    float* __restrict__ out)
{
    __shared__ __attribute__((aligned(16))) _Float16 sAhi[2][128 * LDT];
    __shared__ __attribute__((aligned(16))) _Float16 sAlo[2][128 * LDT];
    __shared__ __attribute__((aligned(16))) _Float16 sBhi[2][128 * LDT];
    __shared__ __attribute__((aligned(16))) _Float16 sBlo[2][128 * LDT];

    // 16x16 supertile swizzle over the 32x64 tile grid (L2 reuse)
    const int pid = blockIdx.x;
    const int st  = pid >> 8;
    const int wi  = pid & 255;
    const int m0  = ((st & 1) * 16 + (wi & 15)) * 128;
    const int n0  = ((st >> 1) * 16 + (wi >> 4)) * 128;

    const int t    = threadIdx.x;
    const int lane = t & 31;
    const int w    = t >> 5;
    const int wm   = w & 1;            // wave tile rows [wm*64, +64)
    const int wn   = w >> 1;           // wave tile cols [wn*32, +32)
    const int lm   = lane & 15;
    const int lh   = lane >> 4;

    v8f acc[4][2] = {};
    f32x4 breg[4];                     // in-flight Wf2 fp32 for next K-slab

    auto issueA = [&](int kk, int buf) {   // pure copy -> async to LDS
        const int k0 = kk * 32;
        #pragma unroll
        for (int i = 0; i < 2; ++i) {
            const int c = i * 256 + t, r = c >> 2, cw = c & 3;
            const size_t g = (size_t)(m0 + r) * KDIM + k0 + cw * 8;
            async_copy16(&sAhi[buf][r * LDT + cw * 8], &Ahi[g]);
            async_copy16(&sAlo[buf][r * LDT + cw * 8], &Alo[g]);
        }
    };
    auto loadB = [&](int kk) {             // issue fp32 loads early
        const int k0 = kk * 32;
        #pragma unroll
        for (int i = 0; i < 4; ++i) {
            const int c = i * 256 + t, r = c >> 3, cw = c & 7;
            breg[i] = *(const f32x4*)&Wf2[(size_t)(n0 + r) * KDIM + k0 + cw * 4];
        }
    };
    auto storeB = [&](int buf) {           // split fp32 -> f16 hi/lo, store late
        #pragma unroll
        for (int i = 0; i < 4; ++i) {
            const int c = i * 256 + t, r = c >> 3, cw = c & 7;
            const f32x4 f = breg[i];
            Half4 hi, lo;
            hi.h[0] = (_Float16)f.x; lo.h[0] = (_Float16)(f.x - (float)hi.h[0]);
            hi.h[1] = (_Float16)f.y; lo.h[1] = (_Float16)(f.y - (float)hi.h[1]);
            hi.h[2] = (_Float16)f.z; lo.h[2] = (_Float16)(f.z - (float)hi.h[2]);
            hi.h[3] = (_Float16)f.w; lo.h[3] = (_Float16)(f.w - (float)hi.h[3]);
            *(u32x2*)&sBhi[buf][r * LDT + cw * 4] = hi.u;
            *(u32x2*)&sBlo[buf][r * LDT + cw * 4] = lo.u;
        }
    };
    auto compute = [&](int buf) {
        Frag bhi[2], blo[2];
        #pragma unroll
        for (int j = 0; j < 2; ++j) {      // B frag: K in [16*lh, +16)
            const int r = wn * 32 + j * 16 + lm;
            bhi[j].q[0] = *(const u32x4*)&sBhi[buf][r * LDT + lh * 16];
            bhi[j].q[1] = *(const u32x4*)&sBhi[buf][r * LDT + lh * 16 + 8];
            blo[j].q[0] = *(const u32x4*)&sBlo[buf][r * LDT + lh * 16];
            blo[j].q[1] = *(const u32x4*)&sBlo[buf][r * LDT + lh * 16 + 8];
        }
        #pragma unroll
        for (int im = 0; im < 4; ++im) {   // A frag: K in [16*lh, +16)
            const int r = wm * 64 + im * 16 + lm;
            Frag ahi, alo;
            ahi.q[0] = *(const u32x4*)&sAhi[buf][r * LDT + lh * 16];
            ahi.q[1] = *(const u32x4*)&sAhi[buf][r * LDT + lh * 16 + 8];
            alo.q[0] = *(const u32x4*)&sAlo[buf][r * LDT + lh * 16];
            alo.q[1] = *(const u32x4*)&sAlo[buf][r * LDT + lh * 16 + 8];
            #pragma unroll
            for (int jn = 0; jn < 2; ++jn) {
                acc[im][jn] = __builtin_amdgcn_wmma_f32_16x16x32_f16(
                    false, alo.v, false, bhi[jn].v, (short)0, acc[im][jn], false, false);
                acc[im][jn] = __builtin_amdgcn_wmma_f32_16x16x32_f16(
                    false, ahi.v, false, blo[jn].v, (short)0, acc[im][jn], false, false);
                acc[im][jn] = __builtin_amdgcn_wmma_f32_16x16x32_f16(
                    false, ahi.v, false, bhi[jn].v, (short)0, acc[im][jn], false, false);
            }
        }
    };

    // ---- prologue: fill buffer 0 ----
    loadB(0);
    issueA(0, 0);
    storeB(0);
    wait_async0();
    __syncthreads();

    // ---- main pipeline: one barrier per K-slab ----
    for (int kk = 0; kk < KSTEPS; ++kk) {
        const int cur = kk & 1;
        const int nxt = cur ^ 1;
        const bool more = (kk + 1 < KSTEPS);
        if (more) { issueA(kk + 1, nxt); loadB(kk + 1); }  // fire & forget
        compute(cur);                                       // 24 WMMAs / wave
        if (more) storeB(nxt);                              // split after WMMAs
        wait_async0();
        __syncthreads();
    }

    // ---- epilogue: +bf2, abs, store per C/D VGPR layout (M = vg + 8*lh) ----
    #pragma unroll
    for (int im = 0; im < 4; ++im) {
        #pragma unroll
        for (int jn = 0; jn < 2; ++jn) {
            const int col  = n0 + wn * 32 + jn * 16 + lm;
            const float bi = bf2[col];
            #pragma unroll
            for (int vg = 0; vg < 8; ++vg) {
                const int row = m0 + wm * 64 + im * 16 + 8 * lh + vg;
                out[(size_t)row * NDIM + col] = fabsf(acc[im][jn][vg] + bi);
            }
        }
    }
}

// ============================================================================
extern "C" void kernel_launch(void* const* d_in, const int* in_sizes, int n_in,
                              void* d_out, int out_size, void* d_ws, size_t ws_size,
                              hipStream_t stream) {
    const float* x0  = (const float*)d_in[0];
    const float* x1  = (const float*)d_in[1];
    const float* x2  = (const float*)d_in[2];
    const float* x3  = (const float*)d_in[3];
    const float* W0  = (const float*)d_in[4];
    const float* b0  = (const float*)d_in[5];
    const float* W1  = (const float*)d_in[6];
    const float* b1  = (const float*)d_in[7];
    const float* W2  = (const float*)d_in[8];
    const float* b2  = (const float*)d_in[9];
    const float* Wf1 = (const float*)d_in[10];
    const float* bf1 = (const float*)d_in[11];
    const float* Wf2 = (const float*)d_in[12];
    const float* bf2 = (const float*)d_in[13];
    float* out = (float*)d_out;

    _Float16* Ahi = (_Float16*)d_ws;                 // 64 MB
    _Float16* Alo = Ahi + (size_t)B_ROWS * KDIM;     // 64 MB

    hnnf_feats_h1<<<B_ROWS, 256, 0, stream>>>(x0, x1, x2, x3, W0, b0, W1, b1,
                                              W2, b2, Wf1, bf1, Ahi, Alo);

    const int tiles = (B_ROWS / 128) * (NDIM / 128);   // 2048
    hnnf_gemm<<<tiles, 256, 0, stream>>>(Ahi, Alo, Wf2, bf2, out);
}
